// VisionEncoder_54065048322466
// MI455X (gfx1250) — matrix operand order repaired and checked by reading, not compiled, not run on hardware
//
#include <hip/hip_runtime.h>
#include <stdint.h>
#include <stddef.h>

// ---------------- problem constants ----------------
#define B_    4
#define L_    1024
#define D_    1024
#define NH_   16
#define HD_   64
#define FFW_  4096
#define LYR_  6
#define PDIM_ 768
#define TOK_  (B_ * L_)        // 4096 tokens
#define EPS_  1e-6f

// ---------------- WMMA / TDM types ----------------
typedef __attribute__((ext_vector_type(16))) __bf16 v16bf;
typedef __attribute__((ext_vector_type(8)))  float  v8f;
typedef __attribute__((ext_vector_type(4)))  unsigned int u32x4;
typedef __attribute__((ext_vector_type(8)))  int          i32x8;
typedef __attribute__((ext_vector_type(4)))  int          i32x4;

union Frag16 { v16bf v; uint4 q[2]; };

__device__ __forceinline__ __bf16 f2b(float f) {
  union { float f; uint32_t u; } a; a.f = f;
  uint32_t r = a.u + 0x7FFFu + ((a.u >> 16) & 1u);   // round-to-nearest-even
  union { uint16_t s; __bf16 b; } o; o.s = (uint16_t)(r >> 16);
  return o.b;
}
__device__ __forceinline__ float b2f(__bf16 b) {
  union { uint16_t s; __bf16 b; } i; i.b = b;
  union { float f; uint32_t u; } o; o.u = ((uint32_t)i.s) << 16;
  return o.f;
}

// Low 32 bits of a flat shared-aperture address = LDS byte offset (ISA 10.2).
__device__ __forceinline__ uint32_t lds_offset(const void* p) {
  return (uint32_t)(uintptr_t)p;
}

// ---------------------------------------------------------------------
// TDM: 2D tile load global -> LDS.
//   tile_k contiguous bf16 elements per row (dim0), tile_rows rows (dim1),
//   memory row stride = stride_elems. LDS receives rows packed contiguously
//   (matches As[rows][tile_k] row-major staging buffer).
// Descriptor packing per CDNA5 ISA section 8.3/8.4 (D# groups 0/1).
// This toolchain exposes the 6-arg builtin form (extra i32x8 group).
// ---------------------------------------------------------------------
__device__ __forceinline__ void tdm_load_2d(uint32_t lds_off, const void* gptr,
                                            uint32_t tile_k, uint32_t tile_rows,
                                            uint64_t stride_elems) {
  uint64_t ga = (uint64_t)(uintptr_t)gptr;
  u32x4 g0;
  g0.x = 1u;                                            // count=1 (valid, user)
  g0.y = lds_off;                                       // lds_addr
  g0.z = (uint32_t)ga;                                  // global_addr[31:0]
  g0.w = (uint32_t)((ga >> 32) & 0x01FFFFFFu) | 0x80000000u;  // addr[56:32], type=2
  uint32_t d0 = tile_k;                                 // tensor_dim0
  uint32_t d1 = tile_rows;                              // tensor_dim1
  uint64_t s0 = stride_elems;                           // tensor_dim0_stride
  i32x8 g1;
  g1[0] = (int)(1u << 16);                              // data_size=1 (2 bytes)
  g1[1] = (int)((d0 & 0xFFFFu) << 16);                  // tensor_dim0[15:0]
  g1[2] = (int)(((d0 >> 16) & 0xFFFFu) | ((d1 & 0xFFFFu) << 16));
  g1[3] = (int)(((d1 >> 16) & 0xFFFFu) | ((tile_k & 0xFFFFu) << 16)); // tile_dim0
  g1[4] = (int)(tile_rows & 0xFFFFu);                   // tile_dim1 (tile_dim2=0)
  g1[5] = (int)(uint32_t)(s0 & 0xFFFFFFFFu);            // stride0[31:0]
  g1[6] = (int)(uint32_t)((s0 >> 32) & 0xFFFFu);        // stride0[47:32]
  g1[7] = 0;
  __builtin_amdgcn_tensor_load_to_lds(g0, g1, (i32x4)0, (i32x4)0, (i32x8)0, 0);
}

// =====================================================================
// Tiled bf16 GEMM:  C[M,N] = A[M,K] * B[K,N], f32 accumulate.
// Block: 256 threads (8 waves), 128x128 tile, K step 32, double-buffered.
// Wave grid 2(M) x 4(N): each wave owns 64x32 = 4x2 WMMA tiles
// (8 v_wmma per K-step per wave, one barrier per K-step).
// A tile staged by the Tensor Data Mover (wave 0 issues, s_wait_tensorcnt);
// B tile staged transposed [N][K] so fragments are contiguous b128 LDS loads.
// =====================================================================
template <bool OUT_BF16>
__global__ __launch_bounds__(256)
void gemm_bf16(const __bf16* __restrict__ A, const __bf16* __restrict__ Bm,
               void* __restrict__ Cv, int M, int N, int K) {
  __shared__ __align__(16) __bf16 As[2][128][32];   // 16 KB
  __shared__ __align__(16) __bf16 Bs[2][128][32];   // 16 KB, [n][k]

  const int tid  = threadIdx.x;
  const int wave = tid >> 5;
  const int lane = tid & 31;
  const int hh   = lane >> 4;
  const int r    = lane & 15;
  const int bm   = blockIdx.y * 128;
  const int bn   = blockIdx.x * 128;
  const int wm   = wave & 1;        // M half (64 rows)
  const int wn   = wave >> 1;       // N quarter (32 cols)

  // B staging: 32 k-rows x 128 n-cols, 16 elems/thread
  const int bk   = tid >> 3;            // 0..31
  const int bn16 = (tid & 7) * 16;      // 0..112

  v8f acc[4][2];
#pragma unroll
  for (int t = 0; t < 4; ++t)
#pragma unroll
    for (int u = 0; u < 2; ++u)
      acc[t][u] = (v8f){0.f,0.f,0.f,0.f,0.f,0.f,0.f,0.f};

  const int nk = K / 32;
  uint4 nbv0, nbv1;

  // ---- prologue: stage tile 0 ----
  if (tid < 32)
    tdm_load_2d(lds_offset(&As[0][0][0]), &A[(size_t)bm * K], 32, 128, (uint64_t)K);
  nbv0 = *(const uint4*)(&Bm[(size_t)bk * N + bn + bn16]);
  nbv1 = *(const uint4*)(&Bm[(size_t)bk * N + bn + bn16 + 8]);
  {
    const __bf16* p0 = (const __bf16*)&nbv0;
    const __bf16* p1 = (const __bf16*)&nbv1;
#pragma unroll
    for (int j = 0; j < 8; ++j) {
      Bs[0][bn16 + j][bk]     = p0[j];
      Bs[0][bn16 + 8 + j][bk] = p1[j];
    }
  }
  if (tid < 32) __builtin_amdgcn_s_wait_tensorcnt(0);
  __syncthreads();

  for (int kt = 0; kt < nk; ++kt) {
    const int cur = kt & 1;
    const int nxt = cur ^ 1;
    const bool more = (kt + 1 < nk);

    // ---- issue next tile's loads early (TDM for A, vector loads for B) ----
    if (more) {
      const int k1 = (kt + 1) * 32;
      if (tid < 32)
        tdm_load_2d(lds_offset(&As[nxt][0][0]), &A[(size_t)bm * K + k1],
                    32, 128, (uint64_t)K);
      nbv0 = *(const uint4*)(&Bm[(size_t)(k1 + bk) * N + bn + bn16]);
      nbv1 = *(const uint4*)(&Bm[(size_t)(k1 + bk) * N + bn + bn16 + 8]);
    }

    // ---- compute on current buffers ----
    Frag16 afr[4], bfr[2];
#pragma unroll
    for (int t = 0; t < 4; ++t) {
      const __bf16* ap = &As[cur][wm * 64 + t * 16 + r][0];
      afr[t].q[0] = *(const uint4*)(ap + 16 * hh);
      afr[t].q[1] = *(const uint4*)(ap + 16 * hh + 8);
    }
#pragma unroll
    for (int u = 0; u < 2; ++u) {
      const __bf16* bp = &Bs[cur][wn * 32 + u * 16 + r][0];
      bfr[u].q[0] = *(const uint4*)(bp + 16 * hh);
      bfr[u].q[1] = *(const uint4*)(bp + 16 * hh + 8);
    }
#pragma unroll
    for (int t = 0; t < 4; ++t)
#pragma unroll
      for (int u = 0; u < 2; ++u)
        acc[t][u] = __builtin_amdgcn_wmma_f32_16x16x32_bf16(
            false, afr[t].v, false, bfr[u].v, (short)0, acc[t][u], false, false);

    // ---- commit next B tile, drain TDM, barrier ----
    if (more) {
      const __bf16* p0 = (const __bf16*)&nbv0;
      const __bf16* p1 = (const __bf16*)&nbv1;
#pragma unroll
      for (int j = 0; j < 8; ++j) {
        Bs[nxt][bn16 + j][bk]     = p0[j];
        Bs[nxt][bn16 + 8 + j][bk] = p1[j];
      }
      if (tid < 32) __builtin_amdgcn_s_wait_tensorcnt(0);
    }
    __syncthreads();
  }

  // ---- epilogue ----
#pragma unroll
  for (int t = 0; t < 4; ++t) {
#pragma unroll
    for (int u = 0; u < 2; ++u) {
#pragma unroll
      for (int v = 0; v < 8; ++v) {
        int row = bm + wm * 64 + t * 16 + v + 8 * hh;
        int col = bn + wn * 32 + u * 16 + r;
        if constexpr (OUT_BF16)
          ((__bf16*)Cv)[(size_t)row * N + col] = f2b(acc[t][u][v]);
        else
          ((float*)Cv)[(size_t)row * N + col] = acc[t][u][v];
      }
    }
  }
}

// =====================================================================
// Fused attention, one (b, head, 16-query block) per workgroup.
// Phase 1: S[16,1024] = Q(16x64) . K^T  via WMMA (contract H=64)
// Phase 2: softmax rows in LDS (bf16 storage, f32 math)
// Phase 3: O[16,64]   = S(16x1024) . V  via WMMA (contract L=1024)
// =====================================================================
__global__ __launch_bounds__(256)
void attn_kernel(const float* __restrict__ qf, const float* __restrict__ kf,
                 const float* __restrict__ vf, __bf16* __restrict__ ob) {
  __shared__ __align__(16) __bf16 Sbuf[16][1024];  // scores / weights (32 KB)
  __shared__ __align__(16) __bf16 Qs[16][64];
  __shared__ __align__(16) __bf16 KVs[64][64];     // K tile [key][h]; V tile [h][key]
  __shared__ float red2[16][17];

  const int q0 = blockIdx.x * 16;
  const int n  = blockIdx.y;
  const int b  = blockIdx.z;
  const int tid  = threadIdx.x;
  const int wave = tid >> 5;
  const int lane = tid & 31;
  const int hh   = lane >> 4;
  const int r    = lane & 15;

  {
    int row = tid >> 4, hc = (tid & 15) * 4;
    const float* src = &qf[(size_t)(b * L_ + q0 + row) * (NH_ * HD_) + n * HD_ + hc];
#pragma unroll
    for (int j = 0; j < 4; ++j) Qs[row][hc + j] = f2b(src[j]);
  }

  // ---- phase 1: scores ----
  for (int k0 = 0; k0 < L_; k0 += 64) {
    int krow = tid >> 2, hc = (tid & 3) * 16;
    const float* src = &kf[(size_t)(b * L_ + k0 + krow) * (NH_ * HD_) + n * HD_ + hc];
#pragma unroll
    for (int j = 0; j < 16; ++j) KVs[krow][hc + j] = f2b(src[j]);
    __syncthreads();
    if (wave < 4) {
      v8f acc = {0.f,0.f,0.f,0.f,0.f,0.f,0.f,0.f};
#pragma unroll
      for (int kk = 0; kk < 64; kk += 32) {
        Frag16 af, bf;
        af.q[0] = *(const uint4*)(&Qs[r][kk + 16 * hh]);
        af.q[1] = *(const uint4*)(&Qs[r][kk + 16 * hh + 8]);
        bf.q[0] = *(const uint4*)(&KVs[wave * 16 + r][kk + 16 * hh]);
        bf.q[1] = *(const uint4*)(&KVs[wave * 16 + r][kk + 16 * hh + 8]);
        acc = __builtin_amdgcn_wmma_f32_16x16x32_bf16(
            false, af.v, false, bf.v, (short)0, acc, false, false);
      }
#pragma unroll
      for (int v = 0; v < 8; ++v)
        Sbuf[v + 8 * hh][k0 + wave * 16 + r] = f2b(acc[v]);
    }
    __syncthreads();
  }

  // ---- phase 2: softmax (16 rows x 16 lanes/row) ----
  {
    int row = tid >> 4, sub = tid & 15;
    float m = -3.4e38f;
    for (int j = sub; j < L_; j += 16) m = fmaxf(m, b2f(Sbuf[row][j]));
    red2[row][sub] = m;
    __syncthreads();
    if (sub == 0) {
      float mm = -3.4e38f;
      for (int t = 0; t < 16; ++t) mm = fmaxf(mm, red2[row][t]);
      red2[row][16] = mm;
    }
    __syncthreads();
    float mm = red2[row][16];
    float s = 0.f;
    for (int j = sub; j < L_; j += 16) {
      float e = __expf(b2f(Sbuf[row][j]) - mm);
      s += e;
      Sbuf[row][j] = f2b(e);
    }
    __syncthreads();
    red2[row][sub] = s;
    __syncthreads();
    if (sub == 0) {
      float ss = 0.f;
      for (int t = 0; t < 16; ++t) ss += red2[row][t];
      red2[row][16] = ss;
    }
    __syncthreads();
    float inv = 1.f / red2[row][16];
    for (int j = sub; j < L_; j += 16)
      Sbuf[row][j] = f2b(b2f(Sbuf[row][j]) * inv);
    __syncthreads();
  }

  // ---- phase 3: O = W . V ----
  v8f oacc = {0.f,0.f,0.f,0.f,0.f,0.f,0.f,0.f};
  for (int k0 = 0; k0 < L_; k0 += 64) {
    int key = tid >> 2, hc = (tid & 3) * 16;
    const float* src = &vf[(size_t)(b * L_ + k0 + key) * (NH_ * HD_) + n * HD_ + hc];
#pragma unroll
    for (int j = 0; j < 16; ++j) KVs[hc + j][key] = f2b(src[j]);  // V^T: [h][key]
    __syncthreads();
    if (wave < 4) {
#pragma unroll
      for (int kk = 0; kk < 64; kk += 32) {
        Frag16 af, bf;
        af.q[0] = *(const uint4*)(&Sbuf[r][k0 + kk + 16 * hh]);
        af.q[1] = *(const uint4*)(&Sbuf[r][k0 + kk + 16 * hh + 8]);
        bf.q[0] = *(const uint4*)(&KVs[wave * 16 + r][kk + 16 * hh]);
        bf.q[1] = *(const uint4*)(&KVs[wave * 16 + r][kk + 16 * hh + 8]);
        oacc = __builtin_amdgcn_wmma_f32_16x16x32_bf16(
            false, af.v, false, bf.v, (short)0, oacc, false, false);
      }
    }
    __syncthreads();
  }
  if (wave < 4) {
#pragma unroll
    for (int v = 0; v < 8; ++v)
      ob[(size_t)(b * L_ + q0 + v + 8 * hh) * (NH_ * HD_) + n * HD_ + wave * 16 + r]
          = f2b(oacc[v]);
  }
}

// =====================================================================
// Elementwise / norm / rope / pooling kernels
// =====================================================================
__global__ void f2b_kernel(const float* __restrict__ s, __bf16* __restrict__ d,
                           long long n) {
  for (long long i = (long long)blockIdx.x * blockDim.x + threadIdx.x; i < n;
       i += (long long)gridDim.x * blockDim.x)
    d[i] = f2b(s[i]);
}

__global__ void pix_prep_kernel(const float* __restrict__ s,
                                __bf16* __restrict__ d, long long n) {
  for (long long i = (long long)blockIdx.x * blockDim.x + threadIdx.x; i < n;
       i += (long long)gridDim.x * blockDim.x)
    d[i] = f2b(2.0f * (s[i] - 0.5f));
}

__global__ void add_pos_kernel(float* __restrict__ x,
                               const float* __restrict__ pos_table,
                               const int* __restrict__ pos) {
  for (long long i = (long long)blockIdx.x * blockDim.x + threadIdx.x;
       i < (long long)TOK_ * D_; i += (long long)gridDim.x * blockDim.x) {
    int token = (int)(i / D_), d = (int)(i % D_);
    int pxr = pos[token * 2], pyr = pos[token * 2 + 1];
    if (pxr == -1 && pyr == -1) continue;  // padded token
    int px = pxr > 0 ? pxr : 0, py = pyr > 0 ? pyr : 0;
    x[i] += pos_table[(size_t)px * D_ + d] +
            pos_table[(size_t)(64 + py) * D_ + d];
  }
}

__global__ __launch_bounds__(256)
void rmsnorm_bf16_kernel(const float* __restrict__ x,
                         const float* __restrict__ scale,
                         __bf16* __restrict__ out) {
  __shared__ float red[256];
  int t = blockIdx.x;
  const float* xr = x + (size_t)t * D_;
  float ss = 0.f;
  for (int j = threadIdx.x; j < D_; j += 256) { float v = xr[j]; ss += v * v; }
  red[threadIdx.x] = ss;
  __syncthreads();
  for (int s = 128; s > 0; s >>= 1) {
    if (threadIdx.x < s) red[threadIdx.x] += red[threadIdx.x + s];
    __syncthreads();
  }
  float rs = rsqrtf(red[0] / (float)D_ + EPS_);
  for (int j = threadIdx.x; j < D_; j += 256)
    out[(size_t)t * D_ + j] = f2b(xr[j] * rs * scale[j]);
}

__global__ __launch_bounds__(256)
void resid_rms_add_kernel(float* __restrict__ x, const float* __restrict__ y,
                          const float* __restrict__ scale) {
  __shared__ float red[256];
  int t = blockIdx.x;
  const float* yr = y + (size_t)t * D_;
  float ss = 0.f;
  for (int j = threadIdx.x; j < D_; j += 256) { float v = yr[j]; ss += v * v; }
  red[threadIdx.x] = ss;
  __syncthreads();
  for (int s = 128; s > 0; s >>= 1) {
    if (threadIdx.x < s) red[threadIdx.x] += red[threadIdx.x + s];
    __syncthreads();
  }
  float rs = rsqrtf(red[0] / (float)D_ + EPS_);
  for (int j = threadIdx.x; j < D_; j += 256)
    x[(size_t)t * D_ + j] += yr[j] * rs * scale[j];
}

// per-(token, head) RMSNorm (+optional scale) and 2D RoPE, in place
__device__ void head_post(float* p, const float* scale, bool do_rope,
                          float px, float py) {
  float buf[HD_];
  float ss = 0.f;
#pragma unroll
  for (int j = 0; j < HD_; ++j) { buf[j] = p[j]; ss += buf[j] * buf[j]; }
  float rs = rsqrtf(ss / (float)HD_ + EPS_);
#pragma unroll
  for (int j = 0; j < HD_; ++j)
    buf[j] = buf[j] * rs * (scale ? scale[j] : 1.0f);
  if (do_rope) {
#pragma unroll
    for (int j = 0; j < 16; ++j) {
      float inv = __powf(10000.0f, -(float)j / 16.0f);
      {
        float ang = px * inv, c = __cosf(ang), s = __sinf(ang);
        float x1 = buf[j], x2 = buf[j + 16];
        buf[j] = x1 * c - x2 * s;
        buf[j + 16] = x2 * c + x1 * s;
      }
      {
        float ang = py * inv, c = __cosf(ang), s = __sinf(ang);
        float x1 = buf[32 + j], x2 = buf[32 + j + 16];
        buf[32 + j] = x1 * c - x2 * s;
        buf[32 + j + 16] = x2 * c + x1 * s;
      }
    }
  }
#pragma unroll
  for (int j = 0; j < HD_; ++j) p[j] = buf[j];
}

__global__ void qkv_post_kernel(float* __restrict__ q, float* __restrict__ k,
                                float* __restrict__ v,
                                const float* __restrict__ qn,
                                const float* __restrict__ kn,
                                const int* __restrict__ pos) {
  int idx = blockIdx.x * blockDim.x + threadIdx.x;  // (token, head)
  if (idx >= TOK_ * NH_) return;
  int token = idx >> 4;
  int n     = idx & 15;
  float px = (float)pos[token * 2];
  float py = (float)pos[token * 2 + 1];
  size_t off = (size_t)token * (NH_ * HD_) + n * HD_;
  head_post(q + off, qn, true,  px, py);
  head_post(k + off, kn, true,  px, py);
  head_post(v + off, nullptr, false, px, py);
}

__global__ void gelu_mul_kernel(const __bf16* __restrict__ g,
                                const __bf16* __restrict__ u,
                                __bf16* __restrict__ o, long long n) {
  for (long long i = (long long)blockIdx.x * blockDim.x + threadIdx.x; i < n;
       i += (long long)gridDim.x * blockDim.x) {
    float gv = b2f(g[i]);
    float t  = 0.7978845608028654f * (gv + 0.044715f * gv * gv * gv);
    float ge = 0.5f * gv * (1.0f + tanhf(t));
    o[i] = f2b(ge * b2f(u[i]));
  }
}

// pooled[b, o, d] = (1/16) * sum_{4x4 patch} x * sqrt(D);  mask = 1
__global__ void pool_kernel(const float* __restrict__ x, float* __restrict__ out) {
  int i = blockIdx.x * blockDim.x + threadIdx.x;
  if (i >= B_ * 64 * D_) return;
  int d = i % D_;
  int o = (i / D_) & 63;
  int b = i / (64 * D_);
  int ox = o & 7, oy = o >> 3;
  float s = 0.f;
#pragma unroll
  for (int jy = 0; jy < 4; ++jy)
#pragma unroll
    for (int jx = 0; jx < 4; ++jx) {
      int l = (ox * 4 + jx) + 32 * (oy * 4 + jy);
      s += x[(size_t)(b * L_ + l) * D_ + d];
    }
  out[i] = s * 2.0f;                 // (1/16) * sqrt(1024)
  if (i < B_ * 64) out[(size_t)B_ * 64 * D_ + i] = 1.0f;  // mask
}

// =====================================================================
// Host-side orchestration
// =====================================================================
static inline void gemm(const __bf16* A, const __bf16* Bm, void* C, int M,
                        int N, int K, bool out_bf16, hipStream_t st) {
  dim3 grid(N / 128, M / 128);
  if (out_bf16)
    gemm_bf16<true><<<grid, 256, 0, st>>>(A, Bm, C, M, N, K);
  else
    gemm_bf16<false><<<grid, 256, 0, st>>>(A, Bm, C, M, N, K);
}

extern "C" void kernel_launch(void* const* d_in, const int* in_sizes, int n_in,
                              void* d_out, int out_size, void* d_ws,
                              size_t ws_size, hipStream_t stream) {
  (void)in_sizes; (void)n_in; (void)out_size; (void)ws_size;

  const float* pix      = (const float*)d_in[0];
  const int*   pos      = (const int*)d_in[1];
  const float* Win      = (const float*)d_in[2];
  const float* pos_tab  = (const float*)d_in[3];
  const float* Wq       = (const float*)d_in[4];
  const float* Wk       = (const float*)d_in[5];
  const float* Wv       = (const float*)d_in[6];
  const float* Wo       = (const float*)d_in[7];
  const float* qn       = (const float*)d_in[8];
  const float* kn       = (const float*)d_in[9];
  const float* pre_attn = (const float*)d_in[10];
  const float* post_attn= (const float*)d_in[11];
  const float* pre_ffw  = (const float*)d_in[12];
  const float* post_ffw = (const float*)d_in[13];
  const float* Wg       = (const float*)d_in[14];
  const float* Wu       = (const float*)d_in[15];
  const float* Wd       = (const float*)d_in[16];
  float* out = (float*)d_out;

  // ---- workspace carve-up ----
  char* ws = (char*)d_ws;
  size_t off = 0;
  auto carve = [&](size_t bytes) {
    char* p = ws + off;
    off += (bytes + 255) & ~(size_t)255;
    return p;
  };
  __bf16* pixb  = (__bf16*)carve((size_t)TOK_ * PDIM_ * 2);
  __bf16* wbuf  = (__bf16*)carve((size_t)D_ * FFW_ * 2);
  float*  x     = (float*) carve((size_t)TOK_ * D_ * 4);
  __bf16* hb    = (__bf16*)carve((size_t)TOK_ * D_ * 2);
  float*  qfb   = (float*) carve((size_t)TOK_ * D_ * 4);
  float*  kfb   = (float*) carve((size_t)TOK_ * D_ * 4);
  float*  vfb   = (float*) carve((size_t)TOK_ * D_ * 4);
  __bf16* ob    = (__bf16*)carve((size_t)TOK_ * D_ * 2);
  float*  oproj = (float*) carve((size_t)TOK_ * D_ * 4);
  __bf16* gb    = (__bf16*)carve((size_t)TOK_ * FFW_ * 2);
  __bf16* ub    = (__bf16*)carve((size_t)TOK_ * FFW_ * 2);
  __bf16* mb    = (__bf16*)carve((size_t)TOK_ * FFW_ * 2);
  float*  mproj = (float*) carve((size_t)TOK_ * D_ * 4);

  const long long nPix = (long long)TOK_ * PDIM_;

  // ---- embed: x = (2*(pix-0.5)) @ Win + pos_emb ----
  pix_prep_kernel<<<4096, 256, 0, stream>>>(pix, pixb, nPix);
  f2b_kernel<<<4096, 256, 0, stream>>>(Win, wbuf, (long long)PDIM_ * D_);
  gemm(pixb, wbuf, x, TOK_, D_, PDIM_, false, stream);
  add_pos_kernel<<<8192, 256, 0, stream>>>(x, pos_tab, pos);

  for (int i = 0; i < LYR_; ++i) {
    const float* Wq_i = Wq + (size_t)i * D_ * D_;
    const float* Wk_i = Wk + (size_t)i * D_ * D_;
    const float* Wv_i = Wv + (size_t)i * D_ * D_;
    const float* Wo_i = Wo + (size_t)i * D_ * D_;
    const float* Wg_i = Wg + (size_t)i * D_ * FFW_;
    const float* Wu_i = Wu + (size_t)i * D_ * FFW_;
    const float* Wd_i = Wd + (size_t)i * FFW_ * D_;

    // ---- attention block ----
    rmsnorm_bf16_kernel<<<TOK_, 256, 0, stream>>>(x, pre_attn + i * D_, hb);

    f2b_kernel<<<4096, 256, 0, stream>>>(Wq_i, wbuf, (long long)D_ * D_);
    gemm(hb, wbuf, qfb, TOK_, D_, D_, false, stream);
    f2b_kernel<<<4096, 256, 0, stream>>>(Wk_i, wbuf, (long long)D_ * D_);
    gemm(hb, wbuf, kfb, TOK_, D_, D_, false, stream);
    f2b_kernel<<<4096, 256, 0, stream>>>(Wv_i, wbuf, (long long)D_ * D_);
    gemm(hb, wbuf, vfb, TOK_, D_, D_, false, stream);

    qkv_post_kernel<<<(TOK_ * NH_) / 256, 256, 0, stream>>>(
        qfb, kfb, vfb, qn + i * HD_, kn + i * HD_, pos);

    attn_kernel<<<dim3(L_ / 16, NH_, B_), 256, 0, stream>>>(qfb, kfb, vfb, ob);

    f2b_kernel<<<4096, 256, 0, stream>>>(Wo_i, wbuf, (long long)D_ * D_);
    gemm(ob, wbuf, oproj, TOK_, D_, D_, false, stream);
    resid_rms_add_kernel<<<TOK_, 256, 0, stream>>>(x, oproj, post_attn + i * D_);

    // ---- FFN block ----
    rmsnorm_bf16_kernel<<<TOK_, 256, 0, stream>>>(x, pre_ffw + i * D_, hb);

    f2b_kernel<<<8192, 256, 0, stream>>>(Wg_i, wbuf, (long long)D_ * FFW_);
    gemm(hb, wbuf, gb, TOK_, FFW_, D_, true, stream);
    f2b_kernel<<<8192, 256, 0, stream>>>(Wu_i, wbuf, (long long)D_ * FFW_);
    gemm(hb, wbuf, ub, TOK_, FFW_, D_, true, stream);

    gelu_mul_kernel<<<8192, 256, 0, stream>>>(gb, ub, mb,
                                              (long long)TOK_ * FFW_);

    f2b_kernel<<<8192, 256, 0, stream>>>(Wd_i, wbuf, (long long)FFW_ * D_);
    gemm(mb, wbuf, mproj, TOK_, D_, FFW_, false, stream);
    resid_rms_add_kernel<<<TOK_, 256, 0, stream>>>(x, mproj, post_ffw + i * D_);
  }

  // ---- pooling + mask ----
  pool_kernel<<<(B_ * 64 * D_) / 256, 256, 0, stream>>>(x, out);
}
